// TriModalSurvival_GNN_23072564314616
// MI455X (gfx1250) — matrix operand order repaired and checked by reading, not compiled, not run on hardware
//
#include <hip/hip_runtime.h>
#include <hip/hip_bf16.h>

// ---------------------------------------------------------------------------
// TriModalSurvival GNN forward for MI455X (gfx1250, wave32, WMMA).
//
// Sizes (match reference): B=8, N=10000, E=250000, LATENT=HID=128.
// n_total = 80000, total edges (incl. self loops) = 2,080,000.
//
// Input pointer layout assumption: depth-first flatten of setup_inputs()
// in dict-insertion order:
//   0 mirna[8,2000] 1 mrna[8,20000] 2 protein[8,10000] 3 edge_weight[250000]
//   4..7   enc_mi  l1.w l1.b l2.w l2.b
//   8..11  enc_mr  l1.w l1.b l2.w l2.b
//   12..15 embed   l1.w l1.b l2.w l2.b
//   16..17 conv0 w,b   18..19 conv1 w,b
//   20..23 bn0 g,b,m,v 24..27 bn1 g,b,m,v
//   28..29 proj w,b
//   30..33 attn l1.w l1.b l2.w l2.b
//   34..35 head.fc1 w,b  36..39 head.bn1 g,b,m,v
//   40..41 head.fc2 w,b  42..45 head.bn2 g,b,m,v
//   46..47 head.fc3 w,b  48..51 head.bn3 g,b,m,v
//   52..53 head.fc4 w,b
//   54 edge_index[2,250000] int32
// Output: (r[8], fused[8*128], attn[8*3]) concatenated = 1056 floats.
// ---------------------------------------------------------------------------

typedef __attribute__((ext_vector_type(16))) __bf16 v16bf;
typedef __attribute__((ext_vector_type(8)))  float  v8f;

#define BQ 8
#define NN 10000
#define EE 250000
#define NT (BQ * NN)        // 80000
#define BE (BQ * EE)        // 2,000,000
#define TOTE (BE + NT)      // 2,080,000
#define HIDD 128
#define EPS 1e-5f

// ---------------------------------------------------------------------------
// bf16 WMMA GEMM: C[M,128] = A[M,K] @ W[K,128] (+bias), fp32 in/out, bf16 MACs.
// M % 16 == 0, K in {64,128}. Block = 256 threads = 8 waves; each wave owns a
// 16-row strip and all 8 column tiles (64 f32 acc VGPRs).
//
// W is staged once per block into LDS, pre-swizzled into WMMA B-fragment
// order [nt][ks][lane][16 bf16], so the inner loop's B read is one contiguous
// 32-byte LDS load. A fragment layout (ISA 7.12.2) is two contiguous 8-float
// runs per lane -> 4 float4 global loads per K-step.
//   A(16x32): row = lane%16; elems 0..7  = A[row, ks + (lane/16)*8 + 0..7]
//                            elems 8..15 = A[row, ks + 16 + (lane/16)*8 + 0..7]
//   B(32x16): col = lane%16; same k order as A (elems 0..7: ks + (lane/16)*8 + i, elems 8..15: ks + 16 + (lane/16)*8 + i-8)
//   C/D(16x16): vgpr r: row = (lane/16)*8 + r; col = lane%16
// ---------------------------------------------------------------------------
__global__ __launch_bounds__(256)
void gemm_bf16_wmma_kernel(const float* __restrict__ A,
                           const float* __restrict__ W,
                           const float* __restrict__ bias,
                           float* __restrict__ C,
                           int M, int K) {
  __shared__ __bf16 wlds[8 * 4 * 32 * 16];     // 32 KB, fragment-ordered W

  const int tid  = threadIdx.x;
  const int ksc  = K >> 5;                     // K-steps (2 or 4)

  // ---- stage W -> LDS (coalesced f32 reads, bf16 fragment-order writes) ----
  for (int e = tid; e < ksc * 4096; e += 256) {
    const int c     = e & 127;                 // output column 0..127
    int r           = e >> 7;
    const int halfl = r & 1;  r >>= 1;
    const int i     = r & 15; r >>= 4;
    const int ks    = r;                       // 0..ksc-1
    const float wv  = W[(long)(ks * 32 + (i & 8) * 2 + halfl * 8 + (i & 7)) * HIDD + c];
    const int nt  = c >> 4, l16 = c & 15;
    wlds[(((nt * 4 + ks) * 32) + halfl * 16 + l16) * 16 + i] = (__bf16)wv;
  }
  __syncthreads();

  const int wave  = tid >> 5;
  const int lane  = tid & 31;
  const int halfl = lane >> 4;
  const int l16   = lane & 15;
  const long mt   = (long)blockIdx.x * 8 + wave;
  if (mt * 16 >= M) return;

  v8f acc[8] = {};
  const float* Arow = A + (mt * 16 + l16) * (long)K + halfl * 8;

  for (int ks = 0; ks < ksc; ++ks) {
    // A fragment: 4 x float4
    const float4 a0 = *(const float4*)(Arow + ks * 32);
    const float4 a1 = *(const float4*)(Arow + ks * 32 + 4);
    const float4 a2 = *(const float4*)(Arow + ks * 32 + 16);
    const float4 a3 = *(const float4*)(Arow + ks * 32 + 20);
    v16bf a;
    a[0] = (__bf16)a0.x;  a[1] = (__bf16)a0.y;  a[2] = (__bf16)a0.z;  a[3] = (__bf16)a0.w;
    a[4] = (__bf16)a1.x;  a[5] = (__bf16)a1.y;  a[6] = (__bf16)a1.z;  a[7] = (__bf16)a1.w;
    a[8] = (__bf16)a2.x;  a[9] = (__bf16)a2.y;  a[10] = (__bf16)a2.z; a[11] = (__bf16)a2.w;
    a[12] = (__bf16)a3.x; a[13] = (__bf16)a3.y; a[14] = (__bf16)a3.z; a[15] = (__bf16)a3.w;
#pragma unroll
    for (int nt = 0; nt < 8; ++nt) {
      const v16bf b = *(const v16bf*)&wlds[(((nt * 4 + ks) * 32) + lane) * 16];
      acc[nt] = __builtin_amdgcn_wmma_f32_16x16x32_bf16(false, a, false, b,
                                                        (short)0, acc[nt],
                                                        false, false);
    }
  }

#pragma unroll
  for (int nt = 0; nt < 8; ++nt) {
    const float bv = bias ? bias[nt * 16 + l16] : 0.0f;
    float* Cp = C + mt * 16 * (long)HIDD + nt * 16 + l16;
#pragma unroll
    for (int r = 0; r < 8; ++r)
      Cp[(long)(halfl * 8 + r) * HIDD] = acc[nt][r] + bv;
  }
}

// ---------------------------------------------------------------------------
// Small dense layer (GEMV-style): Y[m,o] = act_in(X[m,:]) . W[:,o] + bias[o],
// optional fused BatchNorm(eval)+ReLU on the output. Coalesced across o.
// ---------------------------------------------------------------------------
template <int ACT_IN, int FUSE_BN_RELU>
__global__ void dense_kernel(const float* __restrict__ X, int ldx,
                             const float* __restrict__ Wt,
                             const float* __restrict__ bias,
                             const float* __restrict__ bg,
                             const float* __restrict__ bb,
                             const float* __restrict__ bm,
                             const float* __restrict__ bv,
                             float* __restrict__ Y, int ldy,
                             int Mr, int K, int Nc) {
  const int t = blockIdx.x * blockDim.x + threadIdx.x;
  if (t >= Mr * Nc) return;
  const int row = t / Nc, o = t % Nc;
  const float* x = X + (long)row * ldx;
  float acc = bias[o];
  for (int i = 0; i < K; ++i) {
    float xv = x[i];
    if (ACT_IN == 1) xv = xv > 0.f ? xv : 0.01f * xv;   // leaky_relu
    if (ACT_IN == 2) xv = fmaxf(xv, 0.f);               // relu
    acc = fmaf(xv, Wt[(long)i * Nc + o], acc);
  }
  if (FUSE_BN_RELU) {
    acc = (acc - bm[o]) * rsqrtf(bv[o] + EPS) * bg[o] + bb[o];
    acc = fmaxf(acc, 0.f);
  }
  Y[(long)row * ldy + o] = acc;
}

// h1[n,j] = relu(protein[n] * w1[j] + b1[j]),  [NT x 64]
__global__ void embed1_kernel(const float* __restrict__ p,
                              const float* __restrict__ w1,
                              const float* __restrict__ b1,
                              float* __restrict__ h1) {
  const long t = (long)blockIdx.x * blockDim.x + threadIdx.x;
  if (t >= (long)NT * 64) return;
  const long n = t >> 6;
  const int  j = (int)(t & 63);
  h1[t] = fmaxf(fmaf(p[n], w1[j], b1[j]), 0.f);
}

__global__ void deg_init_kernel(float* __restrict__ deg) {
  const int t = blockIdx.x * blockDim.x + threadIdx.x;
  if (t < NT) deg[t] = 1.0f;                 // self-loop weight
}

__global__ void deg_atomic_kernel(const int* __restrict__ ei,
                                  const float* __restrict__ ew,
                                  float* __restrict__ deg) {
  const long t = (long)blockIdx.x * blockDim.x + threadIdx.x;
  if (t >= BE) return;
  const int ee = (int)(t % EE);
  const int b  = (int)(t / EE);
  atomicAdd(&deg[ei[EE + ee] + b * NN], ew[ee]);
}

__global__ void dinv_kernel(const float* __restrict__ deg,
                            float* __restrict__ dinv) {
  const int t = blockIdx.x * blockDim.x + threadIdx.x;
  if (t < NT) { const float d = deg[t]; dinv[t] = d > 0.f ? rsqrtf(d) : 0.f; }
}

__global__ void norm_kernel(const int* __restrict__ ei,
                            const float* __restrict__ ew,
                            const float* __restrict__ dinv,
                            float* __restrict__ nrm) {
  const long t = (long)blockIdx.x * blockDim.x + threadIdx.x;
  if (t >= TOTE) return;
  long s, d; float w;
  if (t < BE) {
    const int ee = (int)(t % EE);
    const int b  = (int)(t / EE);
    s = (long)ei[ee] + (long)b * NN;
    d = (long)ei[EE + ee] + (long)b * NN;
    w = ew[ee];
  } else { s = d = t - BE; w = 1.0f; }
  nrm[t] = dinv[s] * w * dinv[d];
}

// m[dst] += norm_e * hw[src]; 32 lanes per edge, 4 features per lane (float4)
__global__ void scatter_kernel(const int* __restrict__ ei,
                               const float* __restrict__ nrm,
                               const float* __restrict__ hw,
                               float* __restrict__ m) {
  const long t = (long)blockIdx.x * blockDim.x + threadIdx.x;
  const long e = t >> 5;
  if (e >= TOTE) return;
  const int f0 = ((int)t & 31) << 2;
  long s, d;
  if (e < BE) {
    const int ee = (int)(e % EE);
    const int b  = (int)(e / EE);
    s = (long)ei[ee] + (long)b * NN;
    d = (long)ei[EE + ee] + (long)b * NN;
  } else { s = d = e - BE; }
  const float nw = nrm[e];
  const float4 v = *(const float4*)(hw + s * HIDD + f0);
  float* mp = m + d * HIDD + f0;
  atomicAdd(mp + 0, nw * v.x);
  atomicAdd(mp + 1, nw * v.y);
  atomicAdd(mp + 2, nw * v.z);
  atomicAdd(mp + 3, nw * v.w);
}

// h = relu(bn(m + conv_b)) + h   (in place on h)
__global__ void bnres_kernel(const float* __restrict__ m,
                             const float* __restrict__ cb,
                             const float* __restrict__ g,
                             const float* __restrict__ bt,
                             const float* __restrict__ mu,
                             const float* __restrict__ var,
                             float* __restrict__ h) {
  const long t = (long)blockIdx.x * blockDim.x + threadIdx.x;
  if (t >= (long)NT * HIDD) return;
  const int f = (int)(t & (HIDD - 1));
  float x = m[t] + cb[f];
  x = (x - mu[f]) * rsqrtf(var[f] + EPS) * g[f] + bt[f];
  h[t] = fmaxf(x, 0.f) + h[t];
}

// mean pool over nodes into feats[b][1][*] (feats pre-zeroed, atomic partial)
__global__ void pool_kernel(const float* __restrict__ hw,
                            float* __restrict__ feats, int parts) {
  const int b    = blockIdx.x / parts;
  const int part = blockIdx.x % parts;
  const int f    = threadIdx.x & 127;
  const int half = threadIdx.x >> 7;
  float s = 0.f;
  for (int n = part * 2 + half; n < NN; n += parts * 2)
    s += hw[((long)b * NN + n) * HIDD + f];
  __shared__ float red[256];
  red[threadIdx.x] = s;
  __syncthreads();
  if (half == 0)
    atomicAdd(&feats[(b * 3 + 1) * HIDD + f],
              (red[f] + red[128 + f]) / (float)NN);
}

__global__ void softmax_attn_kernel(const float* __restrict__ sc,
                                    float* __restrict__ out_attn) {
  const int b = threadIdx.x;
  if (b >= BQ) return;
  const float s0 = sc[b * 3], s1 = sc[b * 3 + 1], s2 = sc[b * 3 + 2];
  const float mx = fmaxf(s0, fmaxf(s1, s2));
  const float e0 = __expf(s0 - mx), e1 = __expf(s1 - mx), e2 = __expf(s2 - mx);
  const float inv = 1.0f / (e0 + e1 + e2);
  out_attn[b * 3 + 0] = e0 * inv;
  out_attn[b * 3 + 1] = e1 * inv;
  out_attn[b * 3 + 2] = e2 * inv;
}

__global__ void fuse_kernel(const float* __restrict__ feats,
                            const float* __restrict__ attn,
                            float* __restrict__ fused) {
  const int t = blockIdx.x * blockDim.x + threadIdx.x;
  if (t >= BQ * HIDD) return;
  const int b = t >> 7, f = t & 127;
  const float* fb = feats + b * 3 * HIDD;
  fused[t] = attn[b * 3 + 0] * fb[f] +
             attn[b * 3 + 1] * fb[HIDD + f] +
             attn[b * 3 + 2] * fb[2 * HIDD + f];
}

// ---------------------------------------------------------------------------
extern "C" void kernel_launch(void* const* d_in, const int* in_sizes, int n_in,
                              void* d_out, int out_size, void* d_ws, size_t ws_size,
                              hipStream_t stream) {
  (void)in_sizes; (void)n_in; (void)out_size; (void)ws_size;
  auto fin = [&](int i) { return (const float*)d_in[i]; };
  const float* mirna   = fin(0);
  const float* mrna    = fin(1);
  const float* protein = fin(2);
  const float* ew      = fin(3);
  const int*   ei      = (const int*)d_in[54];
  float* out = (float*)d_out;           // [0..8) r, [8..1032) fused, [1032..1056) attn

  // workspace layout (256B aligned)
  char* w = (char*)d_ws;
  auto take = [&](size_t bytes) { char* p = w; w += (bytes + 255) & ~(size_t)255; return p; };
  float* h    = (float*)take((size_t)NT * HIDD * 4);   // 41 MB
  float* hw   = (float*)take((size_t)NT * HIDD * 4);   // 41 MB
  float* m    = (float*)take((size_t)NT * HIDD * 4);   // 41 MB
  float* h1   = (float*)take((size_t)NT * 64 * 4);     // 20.5 MB
  float* nrm  = (float*)take((size_t)TOTE * 4);        // 8.3 MB
  float* deg  = (float*)take((size_t)NT * 4);
  float* dinv = (float*)take((size_t)NT * 4);
  float* feats  = (float*)take((size_t)BQ * 3 * HIDD * 4);
  float* tmp512 = (float*)take((size_t)BQ * 512 * 4);
  float* tattn  = (float*)take((size_t)BQ * 3 * HIDD * 4);
  float* scores = (float*)take((size_t)BQ * 3 * 4);
  float* x2     = (float*)take((size_t)BQ * 256 * 4);
  float* x3     = (float*)take((size_t)BQ * HIDD * 4);

  auto cdiv = [](long a, long b) { return (int)((a + b - 1) / b); };
  const int TB = 256;
  const int gemm_blocks = cdiv((long)NT / 16, 8);      // 8 row-strips per block

  // feats slot 1 accumulated atomically by pool -> zero it up front
  hipMemsetAsync(feats, 0, (size_t)BQ * 3 * HIDD * 4, stream);

  // ---- encoders (write feats slots 0 and 2) ----
  dense_kernel<0, 0><<<cdiv(8 * 512, TB), TB, 0, stream>>>(
      mirna, 2000, fin(4), fin(5), nullptr, nullptr, nullptr, nullptr,
      tmp512, 512, 8, 2000, 512);
  dense_kernel<1, 0><<<cdiv(8 * 128, TB), TB, 0, stream>>>(
      tmp512, 512, fin(6), fin(7), nullptr, nullptr, nullptr, nullptr,
      feats + 0 * HIDD, 3 * HIDD, 8, 512, 128);
  dense_kernel<0, 0><<<cdiv(8 * 512, TB), TB, 0, stream>>>(
      mrna, 20000, fin(8), fin(9), nullptr, nullptr, nullptr, nullptr,
      tmp512, 512, 8, 20000, 512);
  dense_kernel<1, 0><<<cdiv(8 * 128, TB), TB, 0, stream>>>(
      tmp512, 512, fin(10), fin(11), nullptr, nullptr, nullptr, nullptr,
      feats + 2 * HIDD, 3 * HIDD, 8, 512, 128);

  // ---- node embedding: h1 = relu(p*w1+b1); h = h1 @ w2 + b2 (WMMA) ----
  embed1_kernel<<<cdiv((long)NT * 64, TB), TB, 0, stream>>>(protein, fin(12), fin(13), h1);
  gemm_bf16_wmma_kernel<<<gemm_blocks, TB, 0, stream>>>(h1, fin(14), fin(15), h, NT, 64);

  // ---- gcn_norm: deg/dinv/norm (edges + self loops) ----
  deg_init_kernel<<<cdiv(NT, TB), TB, 0, stream>>>(deg);
  deg_atomic_kernel<<<cdiv((long)BE, TB), TB, 0, stream>>>(ei, ew, deg);
  dinv_kernel<<<cdiv(NT, TB), TB, 0, stream>>>(deg, dinv);
  norm_kernel<<<cdiv((long)TOTE, TB), TB, 0, stream>>>(ei, ew, dinv, nrm);

  // ---- two GCN layers: hw = h@W (WMMA); m = scatter(norm*hw); h = relu(bn(m+b))+h
  const int conv_w[2] = {16, 18}, conv_b[2] = {17, 19}, bn0i[2] = {20, 24};
  for (int l = 0; l < 2; ++l) {
    gemm_bf16_wmma_kernel<<<gemm_blocks, TB, 0, stream>>>(
        h, fin(conv_w[l]), nullptr, hw, NT, HIDD);
    hipMemsetAsync(m, 0, (size_t)NT * HIDD * 4, stream);
    scatter_kernel<<<cdiv((long)TOTE * 32, TB), TB, 0, stream>>>(ei, nrm, hw, m);
    bnres_kernel<<<cdiv((long)NT * HIDD, TB), TB, 0, stream>>>(
        m, fin(conv_b[l]), fin(bn0i[l]), fin(bn0i[l] + 1), fin(bn0i[l] + 2),
        fin(bn0i[l] + 3), h);
  }

  // ---- projection + global mean pool -> feats slot 1 ----
  gemm_bf16_wmma_kernel<<<gemm_blocks, TB, 0, stream>>>(
      h, fin(28), fin(29), hw, NT, HIDD);
  const int PARTS = 32;
  pool_kernel<<<BQ * PARTS, TB, 0, stream>>>(hw, feats, PARTS);

  // ---- attention over 3 modalities ----
  dense_kernel<0, 0><<<cdiv(24 * 128, TB), TB, 0, stream>>>(
      feats, 128, fin(30), fin(31), nullptr, nullptr, nullptr, nullptr,
      tattn, 128, 24, 128, 128);
  dense_kernel<1, 0><<<1, TB, 0, stream>>>(
      tattn, 128, fin(32), fin(33), nullptr, nullptr, nullptr, nullptr,
      scores, 1, 24, 128, 1);
  softmax_attn_kernel<<<1, 32, 0, stream>>>(scores, out + 8 + BQ * HIDD);
  fuse_kernel<<<cdiv(BQ * HIDD, TB), TB, 0, stream>>>(feats, out + 8 + BQ * HIDD,
                                                      out + 8);

  // ---- survival head (fc + bn + relu fused) ----
  dense_kernel<0, 1><<<cdiv(8 * 512, TB), TB, 0, stream>>>(
      out + 8, 128, fin(34), fin(35), fin(36), fin(37), fin(38), fin(39),
      tmp512, 512, 8, 128, 512);
  dense_kernel<0, 1><<<cdiv(8 * 256, TB), TB, 0, stream>>>(
      tmp512, 512, fin(40), fin(41), fin(42), fin(43), fin(44), fin(45),
      x2, 256, 8, 512, 256);
  dense_kernel<0, 1><<<cdiv(8 * 128, TB), TB, 0, stream>>>(
      x2, 256, fin(46), fin(47), fin(48), fin(49), fin(50), fin(51),
      x3, 128, 8, 256, 128);
  dense_kernel<0, 0><<<1, TB, 0, stream>>>(
      x3, 128, fin(52), fin(53), nullptr, nullptr, nullptr, nullptr,
      out, 1, 8, 128, 1);
}
